// DynamicOp_695784702117
// MI455X (gfx1250) — matrix operand order repaired and checked by reading, not compiled, not run on hardware
//
#include <hip/hip_runtime.h>
#include <stdint.h>

typedef __attribute__((ext_vector_type(16))) _Float16 v16h;
typedef __attribute__((ext_vector_type(8)))  float    v8f;
typedef __attribute__((ext_vector_type(2)))  float    v2f;

// Sizes fixed by the reference.
#define BB   16
#define CC   64
#define HH_  128
#define WW_  128
#define ZZ   128
#define JTOT (CC*CC*3*3)   // 36864
#define KD   (CC*3*3)      // 576

// ---------------------------------------------------------------------------
// Kernel 1: per-sample weight generation in full FP32 via V_WMMA_F32_16X16X4_F32
//   weight = z @ gen_w + gen_b     ([16 x 128] x [128 x 36864])
// Stored permuted + cast to f16:  Wp[b][o][(kh*3+kw)*64 + c]
// ---------------------------------------------------------------------------
__global__ __launch_bounds__(32) void gen_weights_kernel(
    const float* __restrict__ z, const float* __restrict__ gw,
    const float* __restrict__ gb, _Float16* __restrict__ wp) {
  const int lane = threadIdx.x;
  const int j0   = blockIdx.x * 16;          // 2304 blocks cover 36864 cols
  const int row  = lane & 15;                // M (batch) row / N col sub-index
  const int koff = (lane >> 4) << 1;         // lanes 16-31 handle K+2,K+3

  v8f acc = {};
  for (int k = 0; k < ZZ; k += 4) {
    v2f a, b;
    // 32-bit A 16x4 layout: v0 -> K=0 (lanes0-15) / K=2 (lanes16-31); v1 -> K=1 / K=3
    a.x = z[row * ZZ + k + koff];
    a.y = z[row * ZZ + k + koff + 1];
    // 32-bit B 4x16 layout (mirror of A): v0 -> K=0/2, v1 -> K=1/3, col = lane%16
    b.x = gw[(size_t)(k + koff)     * JTOT + j0 + row];
    b.y = gw[(size_t)(k + koff + 1) * JTOT + j0 + row];
    acc = __builtin_amdgcn_wmma_f32_16x16x4_f32(false, a, false, b,
                                                (short)0, acc, false, false);
  }

  // D layout: VGPR r -> M=r (lanes0-15) / M=8+r (lanes16-31), N=lane%16
  const int j  = j0 + row;                   // flat column in [0,36864)
  const int kw = j % 3, kh = (j / 3) % 3, c = (j / 9) % 64, o = j / 576;
  const int kperm = (kh * 3 + kw) * 64 + c;  // conv-friendly K index
  const float bias = gb[j];
  const int bhi = (lane >> 4) << 3;
#pragma unroll
  for (int r = 0; r < 8; ++r) {
    const int bidx = r + bhi;                // batch sample
    wp[((size_t)(bidx * CC + o)) * KD + kperm] = (_Float16)(acc[r] + bias);
  }
}

// ---------------------------------------------------------------------------
// Kernel 2: per-sample 3x3 SAME conv as implicit GEMM with v_wmma_f32_16x16x32_f16
//   Block tile: M = 64 (all Cout) x N = 128 (one full image row), K = 576 in
//   18 chunks of 32 (chunk t: tap (kh,kw)=t/2, channels c0=(t&1)*32).
//   8 waves = 4 M-groups x 2 N-groups; 4 accumulators (16x64) per wave.
//   A tile staged LDS-side via CDNA5 async DMA (global_load_async_to_lds_b128,
//   ASYNCcnt), overlapping the f32->f16 im2col staging of the B tile.
// ---------------------------------------------------------------------------
__global__ __launch_bounds__(256) void dynconv_kernel(
    const float* __restrict__ x, const _Float16* __restrict__ wp,
    float* __restrict__ out) {
  __shared__ _Float16 As[64 * 32];    // [o][kk]  (kk contiguous, stride 32)
  __shared__ _Float16 Bs[128 * 32];   // [p][kk]  (kk contiguous, stride 32)

  const int tid  = threadIdx.x;
  const int lane = tid & 31;
  const int wv   = tid >> 5;
  const int b    = blockIdx.x >> 7;          // sample
  const int h    = blockIdx.x & 127;         // image row
  const int o0   = (wv >> 1) << 4;           // wave M base: 0,16,32,48
  const int p0   = (wv & 1) << 6;            // wave N base: 0 or 64

  // staging decomposition
  const int sp   = tid & 63;                 // pixel (two passes: sp, sp+64)
  const int skk0 = (tid >> 6) << 3;          // 8 channels per thread
  const int af_  = tid << 3;                 // A staging: 8 halves per thread

  // Async A-tile DMA operands: LDS dest (low 32 bits of generic LDS pointer
  // == wave-relative LDS byte address) + SGPR base / VGPR byte-offset (GVS).
  const uint32_t ldsA    = (uint32_t)(uintptr_t)(&As[af_]);
  const uint64_t wpbase  = (uint64_t)(uintptr_t)wp;
  const uint32_t agoff0  =
      (uint32_t)(((b * CC + (af_ >> 5)) * KD + (af_ & 31)) * 2);

  v8f acc0 = {}, acc1 = {}, acc2 = {}, acc3 = {};

  for (int t = 0; t < 18; ++t) {
    const int khkw = t >> 1;
    const int kh = khkw / 3, kw = khkw % 3;
    const int c0 = (t & 1) << 5;
    const int hh = h + kh - 1;

    // ---- A tile: async DMA 16B/thread straight into LDS (ASYNCcnt)
    {
      const uint32_t goff = agoff0 + (uint32_t)(t * 64);
      asm volatile("global_load_async_to_lds_b128 %0, %1, %2"
                   :: "v"(ldsA), "v"(goff), "s"(wpbase) : "memory");
    }

    // ---- B tile (overlaps the DMA): Bs[p][kk] = f16(x[b,c0+kk,hh,p+kw-1])
    const bool hok = (hh >= 0) & (hh < HH_);
#pragma unroll
    for (int pass = 0; pass < 2; ++pass) {
      const int p  = sp + (pass << 6);
      const int ww = p + kw - 1;
      const bool ok = hok & (ww >= 0) & (ww < WW_);
      _Float16 tmp[8];
#pragma unroll
      for (int q = 0; q < 8; ++q) {
        const int c = c0 + skk0 + q;
        const float v =
            ok ? x[(((size_t)b * CC + c) * HH_ + hh) * WW_ + ww] : 0.0f;
        tmp[q] = (_Float16)v;
      }
      *(uint4*)(&Bs[p * 32 + skk0]) = *(const uint4*)tmp;
    }

    // ---- warm L2/L0 for the next chunk's x slice (global_prefetch_b8)
    if (t < 17) {
      const int khkw2 = (t + 1) >> 1;
      const int kh2 = khkw2 / 3, kw2 = khkw2 % 3;
      const int c02 = ((t + 1) & 1) << 5;
      int hh2 = h + kh2 - 1; hh2 = hh2 < 0 ? 0 : (hh2 > 127 ? 127 : hh2);
      int ww2 = sp + kw2 - 1; ww2 = ww2 < 0 ? 0 : ww2;
      __builtin_prefetch(
          &x[(((size_t)b * CC + c02 + skk0) * HH_ + hh2) * WW_ + ww2], 0, 0);
    }

    asm volatile("s_wait_asynccnt 0" ::: "memory");
    __syncthreads();

    // ---- fragments per ISA 16-bit A/B VGPR layouts, then 4 WMMAs per wave
    union { v16h h; unsigned u[8]; } af, bf;
    const int arow = o0 + (lane & 15);
    const int ahi  = (lane >> 4) << 3;       // A: lanes16-31 -> K+8 per quad
    const int bhi  = (lane >> 4) << 3;       // B: lanes16-31 -> K+8 per quad
#pragma unroll
    for (int v = 0; v < 8; ++v) {
      const int akb = ((v & 3) << 1) + ((v >> 2) << 4) + ahi;
      af.u[v] = *(const unsigned*)(&As[arow * 32 + akb]);
    }
#pragma unroll
    for (int j = 0; j < 4; ++j) {
      const int pc = p0 + (j << 4) + (lane & 15);
#pragma unroll
      for (int v = 0; v < 8; ++v) {
        const int bkb = ((v & 3) << 1) + ((v >> 2) << 4) + bhi;
        bf.u[v] = *(const unsigned*)(&Bs[pc * 32 + bkb]);
      }
      v8f& a = (j == 0) ? acc0 : (j == 1) ? acc1 : (j == 2) ? acc2 : acc3;
      a = __builtin_amdgcn_wmma_f32_16x16x32_f16(false, af.h, false, bf.h,
                                                 (short)0, a, false, false);
    }
    __syncthreads();
  }

  // ---- store D (f32): VGPR r -> M=r / M=8+r, N=lane%16
  const int orow0 = o0 + ((lane >> 4) << 3);
  const int pc_l  = p0 + (lane & 15);
  float* obase = out + ((size_t)b * CC) * (HH_ * WW_) + h * WW_;
#pragma unroll
  for (int r = 0; r < 8; ++r) {
    float* po = obase + (size_t)(orow0 + r) * (HH_ * WW_) + pc_l;
    po[0]  = acc0[r];
    po[16] = acc1[r];
    po[32] = acc2[r];
    po[48] = acc3[r];
  }
}

// ---------------------------------------------------------------------------
extern "C" void kernel_launch(void* const* d_in, const int* in_sizes, int n_in,
                              void* d_out, int out_size, void* d_ws,
                              size_t ws_size, hipStream_t stream) {
  const float* x  = (const float*)d_in[0];
  const float* z  = (const float*)d_in[1];
  const float* gw = (const float*)d_in[2];
  const float* gb = (const float*)d_in[3];
  float* out = (float*)d_out;
  _Float16* wp = (_Float16*)d_ws;   // 16*64*576 halves = 1.15 MB scratch

  gen_weights_kernel<<<JTOT / 16, 32, 0, stream>>>(z, gw, gb, wp);
  dynconv_kernel<<<BB * HH_, 256, 0, stream>>>(x, wp, out);
}
